// Attention_50173807952684
// MI455X (gfx1250) — matrix operand order repaired and checked by reading, not compiled, not run on hardware
//
#include <hip/hip_runtime.h>
#include <hip/hip_bf16.h>
#include <math.h>

#define NTOK 577
#define NPAD 608          // 19 * 32, padded key/row count
#define RT   37           // ceil(577/16) row tiles
#define HEADS 12
#define DH   64
#define EMB  768
#define GG   24
#define EPSV 1e-6f

typedef __bf16 bf16_t;
typedef __attribute__((ext_vector_type(16))) __bf16 v16bf;
typedef __attribute__((ext_vector_type(8)))  float  v8f;

union Frag16 {
    v16bf v;
    uint4 u[2];
    unsigned short s[16];
};

__device__ __forceinline__ unsigned short f2bf(float f) {
    unsigned u = __float_as_uint(f);
    u += 0x7FFFu + ((u >> 16) & 1u);   // round-to-nearest-even
    return (unsigned short)(u >> 16);
}

__device__ __forceinline__ float redmax16(float v) {
    v = fmaxf(v, __shfl_xor(v, 1));
    v = fmaxf(v, __shfl_xor(v, 2));
    v = fmaxf(v, __shfl_xor(v, 4));
    v = fmaxf(v, __shfl_xor(v, 8));
    return v;
}
__device__ __forceinline__ float redsum16(float v) {
    v += __shfl_xor(v, 1);
    v += __shfl_xor(v, 2);
    v += __shfl_xor(v, 4);
    v += __shfl_xor(v, 8);
    return v;
}
__device__ __forceinline__ float softplusf(float x) {
    return x > 20.f ? x : log1pf(__expf(x));
}

// ---------------------------------------------------------------------------
// Kernel 0a: transpose + bf16-convert the three 768x768 weight matrices.
// Output layout Wt[outcol][k] so GEMM B-fragments are contiguous 16-elem runs.
// ---------------------------------------------------------------------------
__global__ void wt_kernel(const float* __restrict__ Wq, const float* __restrict__ Wk,
                          const float* __restrict__ Wv,
                          unsigned short* __restrict__ WtQ, unsigned short* __restrict__ WtK,
                          unsigned short* __restrict__ WtV)
{
    int idx = blockIdx.x * 256 + threadIdx.x;
    if (idx >= EMB * EMB) return;
    int oc = idx / EMB, k = idx - oc * EMB;
    size_t src = (size_t)k * EMB + oc;
    WtQ[idx] = f2bf(Wq[src]);
    WtK[idx] = f2bf(Wk[src]);
    WtV[idx] = f2bf(Wv[src]);
}

// ---------------------------------------------------------------------------
// Kernel 0b: convert x (B*NTOK*EMB f32) to bf16 once, so the GEMM K-loop
// issues raw b128 loads with zero conversion VALU (x was being re-converted
// 12x per row otherwise). 8 elements / thread, b128 in / b128 out.
// ---------------------------------------------------------------------------
__global__ void xb_kernel(const float* __restrict__ x, unsigned short* __restrict__ xb,
                          int total8)
{
    int idx = blockIdx.x * 256 + threadIdx.x;
    if (idx >= total8) return;
    const float4* p = (const float4*)(x + (size_t)idx * 8);
    float4 a = p[0], b = p[1];
    union { uint4 u; unsigned short s[8]; } o;
    o.s[0] = f2bf(a.x); o.s[1] = f2bf(a.y); o.s[2] = f2bf(a.z); o.s[3] = f2bf(a.w);
    o.s[4] = f2bf(b.x); o.s[5] = f2bf(b.y); o.s[6] = f2bf(b.z); o.s[7] = f2bf(b.w);
    ((uint4*)(xb + (size_t)idx * 8))[0] = o.u;
}

// ---------------------------------------------------------------------------
// Kernel 1: fused QKV projection (bf16 WMMA, f32 accum) + var/alpha head.
// grid = (37 row tiles, B, 12 heads); block = 128 (4 waves, 16 cols each).
// Writes: qb/kb bf16 [bh][tok][dh], vT bf16 [bh][dh][tok],
//         alpha / (-0.5/(var+eps)) per token row.
// ---------------------------------------------------------------------------
__global__ __launch_bounds__(128) void qkv_kernel(
    const unsigned short* __restrict__ xb,
    const unsigned short* __restrict__ WtQ,
    const unsigned short* __restrict__ WtK,
    const unsigned short* __restrict__ WtV,
    const float* __restrict__ Wvar, const float* __restrict__ bvar,
    const float* __restrict__ Walpha, const float* __restrict__ balpha,
    unsigned short* __restrict__ qb, unsigned short* __restrict__ kb,
    unsigned short* __restrict__ vT,
    float* __restrict__ alpha_ws, float* __restrict__ nivx_ws, float* __restrict__ nivy_ws)
{
    __shared__ float red[48];   // [16 rows][3: varx, vary, alpha]
    const int tid  = threadIdx.x;
    const int wave = tid >> 5, lane = tid & 31;
    const int half = lane >> 4, l16 = lane & 15;
    const int tile = blockIdx.x, b = blockIdx.y, h = blockIdx.z;
    const int row0 = tile * 16;
    const int bh   = b * HEADS + h;

    if (tid < 48) red[tid] = 0.f;
    __syncthreads();

    const int arow = min(row0 + l16, NTOK - 1);          // clamp pad rows
    const unsigned short* xrow = xb + ((size_t)b * NTOK + arow) * EMB;
    const int col = h * DH + wave * 16 + l16;            // output column
    const unsigned short* wq = WtQ + (size_t)col * EMB;
    const unsigned short* wk = WtK + (size_t)col * EMB;
    const unsigned short* wv = WtV + (size_t)col * EMB;

    v8f cq = {}, ck = {}, cv = {};
    for (int kc = 0; kc < EMB; kc += 32) {
        // prefetch next K-chunks of the weight streams into WGP$
        if (kc + 64 < EMB) {
            __builtin_prefetch(wq + kc + 64, 0, 3);
            __builtin_prefetch(wk + kc + 64, 0, 3);
            __builtin_prefetch(wv + kc + 64, 0, 3);
        }
        // A fragment: two contiguous 16B runs of pre-converted bf16
        Frag16 a;
        a.u[0] = ((const uint4*)(xrow + kc + half * 16))[0];
        a.u[1] = ((const uint4*)(xrow + kc + half * 16 + 8))[0];
        // B fragments: contiguous 16 bf16 (32B) per lane
        const int ko = kc + half * 16;
        Frag16 fq, fk, fv;
        fq.u[0] = ((const uint4*)(wq + ko))[0]; fq.u[1] = ((const uint4*)(wq + ko))[1];
        fk.u[0] = ((const uint4*)(wk + ko))[0]; fk.u[1] = ((const uint4*)(wk + ko))[1];
        fv.u[0] = ((const uint4*)(wv + ko))[0]; fv.u[1] = ((const uint4*)(wv + ko))[1];
        cq = __builtin_amdgcn_wmma_f32_16x16x32_bf16(false, a.v, false, fq.v, (short)0, cq, false, false);
        ck = __builtin_amdgcn_wmma_f32_16x16x32_bf16(false, a.v, false, fk.v, (short)0, ck, false, false);
        cv = __builtin_amdgcn_wmma_f32_16x16x32_bf16(false, a.v, false, fv.v, (short)0, cv, false, false);
    }

    // store q/k row-major, v transposed (token-contiguous)
#pragma unroll
    for (int v = 0; v < 8; ++v) {
        int row = row0 + v + 8 * half;
        if (row < NTOK) {
            size_t qi = ((size_t)bh * NPAD + row) * DH + wave * 16 + l16;
            qb[qi] = f2bf(cq[v]);
            kb[qi] = f2bf(ck[v]);
            vT[((size_t)bh * DH + wave * 16 + l16) * NPAD + row] = f2bf(cv[v]);
        }
    }

    // var / alpha head: dot(q_row, Wvar/Walpha) over dh, cross-wave via LDS atomics
    const int cl = wave * 16 + l16;          // dh index
    const float wv0 = Wvar[cl * 2 + 0];
    const float wv1 = Wvar[cl * 2 + 1];
    const float wa  = Walpha[cl];
#pragma unroll
    for (int v = 0; v < 8; ++v) {
        float p0 = redsum16(cq[v] * wv0);
        float p1 = redsum16(cq[v] * wv1);
        float p2 = redsum16(cq[v] * wa);
        if (l16 == 0) {
            int r = v + 8 * half;
            atomicAdd(&red[r * 3 + 0], p0);
            atomicAdd(&red[r * 3 + 1], p1);
            atomicAdd(&red[r * 3 + 2], p2);
        }
    }
    __syncthreads();
    if (tid < 16) {
        int row = row0 + tid;
        if (row < NTOK) {
            size_t o = (size_t)bh * NPAD + row;
            if (row == 0) {                        // prefix token: no gaussian bias
                alpha_ws[o] = 0.f; nivx_ws[o] = 0.f; nivy_ws[o] = 0.f;
            } else {
                float vx = softplusf(red[tid * 3 + 0] + bvar[0]) + EPSV;
                float vy = softplusf(red[tid * 3 + 1] + bvar[1]) + EPSV;
                float al = softplusf(red[tid * 3 + 2] + balpha[0]);
                alpha_ws[o] = al;
                nivx_ws[o]  = -0.5f / (vx + EPSV);
                nivy_ws[o]  = -0.5f / (vy + EPSV);
            }
        }
    }
}

// ---------------------------------------------------------------------------
// Kernel 2: flash attention with gaussian positional bias.
// grid = (37 row tiles, B, 12 heads); 1 wave per block.
// Streams 32-key chunks: 4 WMMAs for scores, 4 WMMAs for P*V.
// ---------------------------------------------------------------------------
__global__ __launch_bounds__(32) void attn_kernel(
    const unsigned short* __restrict__ qb,
    const unsigned short* __restrict__ kb,
    const unsigned short* __restrict__ vT,
    const float* __restrict__ alpha_ws,
    const float* __restrict__ nivx_ws,
    const float* __restrict__ nivy_ws,
    float* __restrict__ out)
{
    __shared__ unsigned short Plds[16 * 32];     // P tile, row-major [16][32] bf16
    const int lane = threadIdx.x, half = lane >> 4, l16 = lane & 15;
    const int tile = blockIdx.x, b = blockIdx.y, h = blockIdx.z;
    const int row0 = tile * 16, bh = b * HEADS + h;

    // preload Q A-fragments for both dh 32-chunks
    const unsigned short* qrow = qb + ((size_t)bh * NPAD + row0 + l16) * DH;
    Frag16 qa[2];
#pragma unroll
    for (int d = 0; d < 2; ++d) {
        qa[d].u[0] = ((const uint4*)(qrow + d * 32 + half * 16))[0];
        qa[d].u[1] = ((const uint4*)(qrow + d * 32 + half * 16 + 8))[0];
    }

    float alp[8], nvx[8], nvy[8], grf[8], gcf[8], m8[8], s8[8];
#pragma unroll
    for (int v = 0; v < 8; ++v) {
        int row = row0 + v + 8 * half;
        size_t o = (size_t)bh * NPAD + row;
        alp[v] = alpha_ws[o]; nvx[v] = nivx_ws[o]; nvy[v] = nivy_ws[o];
        int i = row - 1; if (i < 0) i = 0;
        grf[v] = (float)(i / GG); gcf[v] = (float)(i % GG);
        m8[v] = -__builtin_inff(); s8[v] = 0.f;
    }
    v8f oacc[4];
#pragma unroll
    for (int dc = 0; dc < 4; ++dc) oacc[dc] = (v8f){};

    for (int kc = 0; kc < NPAD; kc += 32) {
        // prefetch next chunk's K rows into WGP$
        if (kc + 32 < NPAD) {
            __builtin_prefetch(kb + ((size_t)bh * NPAD + kc + 32 + lane) * DH, 0, 3);
        }
        float sf[2][8];
#pragma unroll
        for (int n = 0; n < 2; ++n) {
            const unsigned short* krow = kb + ((size_t)bh * NPAD + kc + n * 16 + l16) * DH;
            Frag16 kf0, kf1;
            kf0.u[0] = ((const uint4*)(krow + half * 16))[0];
            kf0.u[1] = ((const uint4*)(krow + half * 16))[1];
            kf1.u[0] = ((const uint4*)(krow + 32 + half * 16))[0];
            kf1.u[1] = ((const uint4*)(krow + 32 + half * 16))[1];
            v8f s = {};
            s = __builtin_amdgcn_wmma_f32_16x16x32_bf16(false, qa[0].v, false, kf0.v, (short)0, s, false, false);
            s = __builtin_amdgcn_wmma_f32_16x16x32_bf16(false, qa[1].v, false, kf1.v, (short)0, s, false, false);
            int j = kc + n * 16 + l16;            // global key index (col of this lane)
            bool jval = (j >= 1) && (j < NTOK);
            int jj = jval ? j - 1 : 0;
            float gjr = (float)(jj / GG), gjc = (float)(jj % GG);
#pragma unroll
            for (int v = 0; v < 8; ++v) {
                float sc = s[v] * 0.125f;          // 1/sqrt(64)
                if (j < NTOK) {
                    if (jval) {
                        float dx = grf[v] - gjr, dy = gcf[v] - gjc;
                        sc += alp[v] * __expf(dx * dx * nvx[v] + dy * dy * nvy[v]);
                    }
                } else {
                    sc = -__builtin_inff();        // pad keys
                }
                sf[n][v] = sc;
            }
        }
        // online softmax: per-row max across 32 cols (16 lanes x 2 subtiles)
        float scl[8];
#pragma unroll
        for (int v = 0; v < 8; ++v) {
            float rm = redmax16(fmaxf(sf[0][v], sf[1][v]));
            float nm = fmaxf(m8[v], rm);
            float sc = __expf(m8[v] - nm);
            m8[v] = nm; scl[v] = sc; s8[v] *= sc;
        }
#pragma unroll
        for (int v = 0; v < 8; ++v) {
            oacc[0][v] *= scl[v]; oacc[1][v] *= scl[v];
            oacc[2][v] *= scl[v]; oacc[3][v] *= scl[v];
        }
#pragma unroll
        for (int v = 0; v < 8; ++v) {
            float p0 = __expf(sf[0][v] - m8[v]);
            float p1 = __expf(sf[1][v] - m8[v]);
            s8[v] += redsum16(p0 + p1);
            int r = v + 8 * half;                  // C-layout row
            Plds[r * 32 + l16]      = f2bf(p0);
            Plds[r * 32 + 16 + l16] = f2bf(p1);
        }
        __syncthreads();
        // reload P in A-layout from LDS
        Frag16 pa;
        pa.u[0] = ((const uint4*)(Plds + l16 * 32 + half * 16))[0];
        pa.u[1] = ((const uint4*)(Plds + l16 * 32 + half * 16 + 8))[0];
        // accumulate P * V over 4 dh column tiles (vT token-contiguous)
#pragma unroll
        for (int dc = 0; dc < 4; ++dc) {
            const unsigned short* vrow =
                vT + ((size_t)bh * DH + dc * 16 + l16) * NPAD + kc + half * 16;
            Frag16 vf;
            vf.u[0] = ((const uint4*)vrow)[0];
            vf.u[1] = ((const uint4*)vrow)[1];
            oacc[dc] = __builtin_amdgcn_wmma_f32_16x16x32_bf16(
                false, pa.v, false, vf.v, (short)0, oacc[dc], false, false);
        }
        __syncthreads();
    }

    // normalize + store (B, N, E) f32
#pragma unroll
    for (int v = 0; v < 8; ++v) {
        int row = row0 + v + 8 * half;
        if (row < NTOK) {
            float inv = 1.f / s8[v];
            size_t base = ((size_t)b * NTOK + row) * EMB + h * DH;
            out[base + 0  + l16] = oacc[0][v] * inv;
            out[base + 16 + l16] = oacc[1][v] * inv;
            out[base + 32 + l16] = oacc[2][v] * inv;
            out[base + 48 + l16] = oacc[3][v] * inv;
        }
    }
}

// ---------------------------------------------------------------------------
extern "C" void kernel_launch(void* const* d_in, const int* in_sizes, int n_in,
                              void* d_out, int out_size, void* d_ws, size_t ws_size,
                              hipStream_t stream) {
    (void)n_in; (void)out_size; (void)ws_size;
    const float* x      = (const float*)d_in[0];
    const float* Wq     = (const float*)d_in[1];
    const float* Wk     = (const float*)d_in[2];
    const float* Wv     = (const float*)d_in[3];
    const float* Wvar   = (const float*)d_in[4];
    const float* bvar   = (const float*)d_in[5];
    const float* Walpha = (const float*)d_in[6];
    const float* balpha = (const float*)d_in[7];
    float* out = (float*)d_out;
    const int B = in_sizes[0] / (NTOK * EMB);

    char* ws = (char*)d_ws;
    size_t off = 0;
    auto alloc = [&](size_t bytes) -> char* {
        char* p = ws + off;
        off = (off + bytes + 255) & ~(size_t)255;
        return p;
    };
    unsigned short* WtQ = (unsigned short*)alloc((size_t)EMB * EMB * 2);
    unsigned short* WtK = (unsigned short*)alloc((size_t)EMB * EMB * 2);
    unsigned short* WtV = (unsigned short*)alloc((size_t)EMB * EMB * 2);
    unsigned short* xbb = (unsigned short*)alloc((size_t)B * NTOK * EMB * 2);
    size_t tokbuf = (size_t)B * HEADS * NPAD * DH * 2;
    unsigned short* qb = (unsigned short*)alloc(tokbuf);
    unsigned short* kb = (unsigned short*)alloc(tokbuf);
    unsigned short* vT = (unsigned short*)alloc(tokbuf);
    size_t rowbuf = (size_t)B * HEADS * NPAD * 4;
    float* alpha_ws = (float*)alloc(rowbuf);
    float* nivx_ws  = (float*)alloc(rowbuf);
    float* nivy_ws  = (float*)alloc(rowbuf);

    wt_kernel<<<(EMB * EMB + 255) / 256, 256, 0, stream>>>(Wq, Wk, Wv, WtQ, WtK, WtV);
    const int total8 = (B * NTOK * EMB) / 8;     // E=768 divisible by 8
    xb_kernel<<<(total8 + 255) / 256, 256, 0, stream>>>(x, xbb, total8);
    qkv_kernel<<<dim3(RT, B, HEADS), 128, 0, stream>>>(
        xbb, WtQ, WtK, WtV, Wvar, bvar, Walpha, balpha,
        qb, kb, vT, alpha_ws, nivx_ws, nivy_ws);
    attn_kernel<<<dim3(RT, B, HEADS), 32, 0, stream>>>(
        qb, kb, vT, alpha_ws, nivx_ws, nivy_ws, out);
}
